// GemmaAttention_63788854280216
// MI455X (gfx1250) — matrix operand order repaired and checked by reading, not compiled, not run on hardware
//
#include <hip/hip_runtime.h>

typedef _Float16 v16h __attribute__((ext_vector_type(16)));
typedef float    v8f  __attribute__((ext_vector_type(8)));
typedef unsigned int v4u __attribute__((ext_vector_type(4)));

union Frag { v4u q[2]; v16h v; };

constexpr int B_  = 2;
constexpr int S_  = 2048;
constexpr int D_  = 2048;
constexpr int H_  = 8;
constexpr int HD_ = 256;
constexpr float SCALE_ = 0.0625f;   // 1/sqrt(256)

__device__ __forceinline__ v8f wmma16(v16h a, v16h b, v8f c) {
    // D = A(16x32 f16) * B(32x16 f16) + C(16x16 f32)
    return __builtin_amdgcn_wmma_f32_16x16x32_f16(
        false, a, false, b, (short)0, c, false, false);
}

__device__ __forceinline__ v8f zero8() {
    v8f z = {0.f,0.f,0.f,0.f,0.f,0.f,0.f,0.f};
    return z;
}

// ---------------------------------------------------------------- convert
__global__ void conv_f16(const float* __restrict__ src, _Float16* __restrict__ dst, int n) {
    int i = blockIdx.x * blockDim.x + threadIdx.x;
    if (i < n) dst[i] = (_Float16)src[i];
}

// transpose + convert: src (K x N) f32 row-major -> dst (N x K) f16 row-major
// block (32,8); grid (N/32, K/32, Z); Z batches offset by K*N elements each.
__global__ __launch_bounds__(256) void transpose_f16(
    const float* __restrict__ src, _Float16* __restrict__ dst, int K, int N) {
    __shared__ float tile[32][33];
    const int tx = threadIdx.x, ty = threadIdx.y;
    const int n0 = blockIdx.x * 32, k0 = blockIdx.y * 32;
    const size_t zoff = (size_t)blockIdx.z * (size_t)K * (size_t)N;
    src += zoff; dst += zoff;
#pragma unroll
    for (int i = 0; i < 4; ++i)
        tile[ty + i*8][tx] = src[(size_t)(k0 + ty + i*8) * N + n0 + tx];
    __syncthreads();
#pragma unroll
    for (int i = 0; i < 4; ++i)
        dst[(size_t)(n0 + ty + i*8) * K + k0 + tx] = (_Float16)tile[tx][ty + i*8];
}

// ---------------------------------------------------------------- GEMM
// C(MxN,f32) = A(MxK,f16,row-major) * Bt(NxK,f16,row-major == B transposed)
// 256 threads = 8 waves; block tile 128x64; wave tile 32x32; BK = 32.
__global__ __launch_bounds__(256) void gemm_f16(
    const _Float16* __restrict__ A, const _Float16* __restrict__ Bt,
    float* __restrict__ C, int M, int N, int K) {
    __shared__ _Float16 As[128][32];
    __shared__ _Float16 Bs[64][32];
    const int tid  = threadIdx.x;
    const int wave = tid >> 5, lane = tid & 31;
    const int l16  = lane & 15, lh = lane >> 4;
    const int m0 = blockIdx.y * 128;
    const int n0 = blockIdx.x * 64;
    const int mt = (wave & 3) * 32;
    const int nt = (wave >> 2) * 32;
    const int kb = lh * 16;

    v8f acc00 = zero8(), acc01 = zero8(), acc10 = zero8(), acc11 = zero8();

    const int arow = tid >> 1, aseg = tid & 1;   // A stage: 128 rows x 32 halves
    const int brow = tid >> 2, bseg = tid & 3;   // B stage: 64 rows x 32 halves

    for (int k0 = 0; k0 < K; k0 += 32) {
        const v4u* asrc = (const v4u*)(A + (size_t)(m0 + arow) * K + k0 + aseg * 16);
        v4u* adst = (v4u*)&As[arow][aseg * 16];
        adst[0] = asrc[0];
        adst[1] = asrc[1];
        *(v4u*)&Bs[brow][bseg * 8] =
            *(const v4u*)(Bt + (size_t)(n0 + brow) * K + k0 + bseg * 8);
        __syncthreads();

        Frag a0, a1, b0, b1;
        a0.q[0] = *(const v4u*)&As[mt + l16][kb];
        a0.q[1] = *(const v4u*)&As[mt + l16][kb + 8];
        a1.q[0] = *(const v4u*)&As[mt + 16 + l16][kb];
        a1.q[1] = *(const v4u*)&As[mt + 16 + l16][kb + 8];
        b0.q[0] = *(const v4u*)&Bs[nt + l16][lh * 16];
        b0.q[1] = *(const v4u*)&Bs[nt + l16][lh * 16 + 8];
        b1.q[0] = *(const v4u*)&Bs[nt + 16 + l16][lh * 16];
        b1.q[1] = *(const v4u*)&Bs[nt + 16 + l16][lh * 16 + 8];

        acc00 = wmma16(a0.v, b0.v, acc00);
        acc01 = wmma16(a0.v, b1.v, acc01);
        acc10 = wmma16(a1.v, b0.v, acc10);
        acc11 = wmma16(a1.v, b1.v, acc11);
        __syncthreads();
    }
    // C layout: vgpr r -> row lh*8+r, col = l16
#pragma unroll
    for (int r = 0; r < 8; ++r) {
        size_t row0 = (size_t)(m0 + mt + lh * 8 + r) * N + n0 + nt;
        size_t row1 = (size_t)(m0 + mt + 16 + lh * 8 + r) * N + n0 + nt;
        C[row0 + l16]      = acc00[r];
        C[row0 + 16 + l16] = acc01[r];
        C[row1 + l16]      = acc10[r];
        C[row1 + 16 + l16] = acc11[r];
    }
}

// ---------------------------------------------------------------- RoPE
__global__ void rope_q(const float* __restrict__ qf, const int* __restrict__ pos,
                       _Float16* __restrict__ qh) {
    int i = blockIdx.x * blockDim.x + threadIdx.x;   // B*S*H*128 = 2^22
    int d = i & 127;
    int h = (i >> 7) & 7;
    int s = (i >> 10) & 2047;
    int b = i >> 21;
    float p    = (float)pos[b * S_ + s];
    float invf = __powf(10000.0f, -(float)d * (1.0f / 128.0f));
    float ang  = p * invf;
    float cs = __cosf(ang), sn = __sinf(ang);
    const float* row = qf + ((size_t)(b * S_ + s)) * (H_ * HD_) + h * HD_;
    float x1 = row[d], x2 = row[d + 128];
    _Float16* out = qh + (((size_t)(b * H_ + h)) * S_ + s) * HD_;
    out[d]       = (_Float16)(x1 * cs - x2 * sn);
    out[d + 128] = (_Float16)(x2 * cs + x1 * sn);
}

__global__ void rope_k(const float* __restrict__ kf, const int* __restrict__ pos,
                       _Float16* __restrict__ kh) {
    int i = blockIdx.x * blockDim.x + threadIdx.x;   // B*S*128 = 2^19
    int d = i & 127;
    int s = (i >> 7) & 2047;
    int b = i >> 18;
    float p    = (float)pos[b * S_ + s];
    float invf = __powf(10000.0f, -(float)d * (1.0f / 128.0f));
    float ang  = p * invf;
    float cs = __cosf(ang), sn = __sinf(ang);
    const float* row = kf + (size_t)(b * S_ + s) * HD_;
    float x1 = row[d], x2 = row[d + 128];
    _Float16* out = kh + (size_t)(b * S_ + s) * HD_;
    out[d]       = (_Float16)(x1 * cs - x2 * sn);
    out[d + 128] = (_Float16)(x2 * cs + x1 * sn);
}

// ---------------------------------------------------------------- flash attention
// grid (S/64, H, B), block = 128 (4 waves). Wave owns 16 q rows.
// qh [B,H,S,HD] f16; kh [B,S,HD] f16; vT [B,HD,S] f16; mask [B,1,S,S] f32
// outh [B,S,H*HD] f16
__global__ __launch_bounds__(128) void attn_kernel(
    const _Float16* __restrict__ qh, const _Float16* __restrict__ kh,
    const _Float16* __restrict__ vT, const float* __restrict__ mask,
    _Float16* __restrict__ outh) {
    __shared__ _Float16 Ks[32][HD_];     // 16 KB : 32 keys x 256
    __shared__ _Float16 Vs[HD_][32];     // 16 KB : 256 hd x 32 keys
    __shared__ _Float16 Ps[4][16][32];   //  4 KB : per-wave P tile

    const int h    = blockIdx.y;
    const int b    = blockIdx.z;
    const int tid  = threadIdx.x;
    const int wave = tid >> 5;
    const int lane = tid & 31;
    const int l16  = lane & 15;
    const int lh   = lane >> 4;
    const int kb   = lh * 16;
    const int q0   = blockIdx.x * 64 + wave * 16;   // wave's q-row base

    // Q fragments held in registers: 8 chunks of K=32 over HD=256
    const _Float16* qbase = qh + (((size_t)(b * H_ + h)) * S_ + q0 + l16) * HD_;
    Frag aq[8];
#pragma unroll
    for (int c = 0; c < 8; ++c) {
        aq[c].q[0] = *(const v4u*)(qbase + c * 32 + kb);
        aq[c].q[1] = *(const v4u*)(qbase + c * 32 + kb + 8);
    }

    v8f o[16];
#pragma unroll
    for (int c = 0; c < 16; ++c) o[c] = zero8();
    float m[8], l[8];
#pragma unroll
    for (int r = 0; r < 8; ++r) { m[r] = -1e30f; l[r] = 0.0f; }

    const float* maskbase = mask + ((size_t)b * S_ + q0) * S_;

    for (int k0 = 0; k0 < S_; k0 += 32) {
        // ---- stage K tile (contiguous 16KB) and V tile (transposed rows)
        {
            const v4u* src = (const v4u*)(kh + ((size_t)b * S_ + k0) * HD_);
            v4u* dst = (v4u*)&Ks[0][0];
#pragma unroll
            for (int i = 0; i < 8; ++i) dst[tid + i * 128] = src[tid + i * 128];
#pragma unroll
            for (int i = 0; i < 8; ++i) {
                int idx = tid + i * 128;    // 0..1023
                int hd  = idx >> 2;
                int seg = idx & 3;
                ((v4u*)&Vs[hd][0])[seg] =
                    ((const v4u*)(vT + ((size_t)b * HD_ + hd) * S_ + k0))[seg];
            }
        }
        __syncthreads();

        // ---- scores: 2 x (16x16) tiles, reduce over HD with 8 WMMA each
        v8f c0 = zero8(), c1 = zero8();
#pragma unroll
        for (int c = 0; c < 8; ++c) {
            Frag bk0, bk1;
            const _Float16* kr0 = &Ks[l16][c * 32 + lh * 16];
            const _Float16* kr1 = &Ks[16 + l16][c * 32 + lh * 16];
            bk0.q[0] = *(const v4u*)(kr0);
            bk0.q[1] = *(const v4u*)(kr0 + 8);
            bk1.q[0] = *(const v4u*)(kr1);
            bk1.q[1] = *(const v4u*)(kr1 + 8);
            c0 = wmma16(aq[c].v, bk0.v, c0);
            c1 = wmma16(aq[c].v, bk1.v, c1);
        }

        // ---- online softmax (rows: lh*8+r; cols across 16-lane group)
        float alpha[8];
#pragma unroll
        for (int r = 0; r < 8; ++r) {
            size_t mrow = (size_t)(lh * 8 + r) * S_ + k0;
            float s0 = c0[r] * SCALE_ + maskbase[mrow + l16];
            float s1 = c1[r] * SCALE_ + maskbase[mrow + 16 + l16];
            float t = fmaxf(s0, s1);
            for (int off = 8; off; off >>= 1) t = fmaxf(t, __shfl_xor(t, off, 16));
            float mn = fmaxf(m[r], t);
            alpha[r] = __expf(m[r] - mn);
            m[r] = mn;
            float p0 = __expf(s0 - mn);
            float p1 = __expf(s1 - mn);
            float rs = p0 + p1;
            for (int off = 8; off; off >>= 1) rs += __shfl_xor(rs, off, 16);
            l[r] = l[r] * alpha[r] + rs;
            Ps[wave][lh * 8 + r][l16]      = (_Float16)p0;
            Ps[wave][lh * 8 + r][16 + l16] = (_Float16)p1;
        }
        // rescale output accumulators
#pragma unroll
        for (int c = 0; c < 16; ++c)
#pragma unroll
            for (int r = 0; r < 8; ++r) o[c][r] *= alpha[r];

        // per-wave LDS fence: P stores must land before fragment reload
        asm volatile("s_wait_dscnt 0" ::: "memory");

        Frag ap;
        ap.q[0] = *(const v4u*)&Ps[wave][l16][kb];
        ap.q[1] = *(const v4u*)&Ps[wave][l16][kb + 8];

        // ---- O += P(16x32) @ V(32x256): 16 WMMA over HD chunks
#pragma unroll
        for (int c = 0; c < 16; ++c) {
            Frag bv;
            const _Float16* vr = &Vs[c * 16 + l16][lh * 16];
            bv.q[0] = *(const v4u*)(vr);
            bv.q[1] = *(const v4u*)(vr + 8);
            o[c] = wmma16(ap.v, bv.v, o[c]);
        }
        __syncthreads();
    }

    // ---- epilogue: normalize and write f16 [B,S,H*HD]
    _Float16* ob = outh + ((size_t)b * S_ + q0) * (H_ * HD_) + h * HD_;
#pragma unroll
    for (int c = 0; c < 16; ++c)
#pragma unroll
        for (int r = 0; r < 8; ++r) {
            float val = o[c][r] / l[r];
            ob[(size_t)(lh * 8 + r) * (H_ * HD_) + c * 16 + l16] = (_Float16)val;
        }
}

// ---------------------------------------------------------------- launch
extern "C" void kernel_launch(void* const* d_in, const int* in_sizes, int n_in,
                              void* d_out, int out_size, void* d_ws, size_t ws_size,
                              hipStream_t stream) {
    const float* hs   = (const float*)d_in[0];
    const float* mask = (const float*)d_in[1];
    const int*   pos  = (const int*)d_in[2];
    const float* Wq   = (const float*)d_in[3];
    const float* Wk   = (const float*)d_in[4];
    const float* Wv   = (const float*)d_in[5];
    const float* Wo   = (const float*)d_in[6];
    float* out = (float*)d_out;

    char* ws = (char*)d_ws;
    const size_t BS = (size_t)B_ * S_;                 // 4096
    size_t off = 0;
    _Float16* Xh  = (_Float16*)(ws + off); off += BS * D_ * 2;          // X f16
    _Float16* WqT = (_Float16*)(ws + off); off += (size_t)D_ * D_ * 2;  // Wq^T
    _Float16* WkT = (_Float16*)(ws + off); off += (size_t)HD_ * D_ * 2; // Wk^T
    _Float16* WvT = (_Float16*)(ws + off); off += (size_t)HD_ * D_ * 2; // Wv^T
    _Float16* WoT = (_Float16*)(ws + off); off += (size_t)D_ * D_ * 2;  // Wo^T
    float*    Qf  = (float*)(ws + off);    off += BS * (size_t)(H_*HD_) * 4;
    float*    Kf  = (float*)(ws + off);    off += BS * (size_t)HD_ * 4;
    float*    Vf  = (float*)(ws + off);    off += BS * (size_t)HD_ * 4;
    _Float16* Qh  = (_Float16*)(ws + off); off += BS * (size_t)(H_*HD_) * 2;
    _Float16* Kh  = (_Float16*)(ws + off); off += BS * (size_t)HD_ * 2;
    _Float16* Vt  = (_Float16*)(ws + off); off += BS * (size_t)HD_ * 2;
    _Float16* Ah  = (_Float16*)(ws + off); off += BS * (size_t)(H_*HD_) * 2;

    dim3 tb(32, 8);
    // f16 copies of activations / transposed weights
    conv_f16<<<(int)(BS * D_ / 256), 256, 0, stream>>>(hs, Xh, (int)(BS * D_));
    transpose_f16<<<dim3(64, 64, 1), tb, 0, stream>>>(Wq, WqT, D_, D_);
    transpose_f16<<<dim3(8, 64, 1),  tb, 0, stream>>>(Wk, WkT, D_, HD_);
    transpose_f16<<<dim3(8, 64, 1),  tb, 0, stream>>>(Wv, WvT, D_, HD_);
    transpose_f16<<<dim3(64, 64, 1), tb, 0, stream>>>(Wo, WoT, D_, D_);
    // QKV projections (WMMA)
    gemm_f16<<<dim3(32, 32), 256, 0, stream>>>(Xh, WqT, Qf, 4096, 2048, 2048);
    gemm_f16<<<dim3(4, 32),  256, 0, stream>>>(Xh, WkT, Kf, 4096, 256, 2048);
    gemm_f16<<<dim3(4, 32),  256, 0, stream>>>(Xh, WvT, Vf, 4096, 256, 2048);
    // RoPE + precision/layout conversion
    rope_q<<<16384, 256, 0, stream>>>(Qf, pos, Qh);
    rope_k<<<2048, 256, 0, stream>>>(Kf, pos, Kh);
    transpose_f16<<<dim3(8, 64, 2), tb, 0, stream>>>(Vf, Vt, S_, HD_); // V -> [B,HD,S]
    // flash attention (WMMA scores + WMMA AV, online softmax)
    attn_kernel<<<dim3(32, 8, 2), 128, 0, stream>>>(Qh, Kh, Vt, mask, Ah);
    // output projection (WMMA)
    gemm_f16<<<dim3(32, 32), 256, 0, stream>>>(Ah, WoT, out, 4096, 2048, 2048);
}
